// Sparse_Graph_32315333935699
// MI455X (gfx1250) — matrix operand order repaired and checked by reading, not compiled, not run on hardware
//
#include <hip/hip_runtime.h>
#include <hip/hip_bf16.h>
#include <math.h>

typedef __attribute__((ext_vector_type(16))) _Float16 v16h;
typedef __attribute__((ext_vector_type(8)))  _Float16 v8h;
typedef __attribute__((ext_vector_type(8)))  float    v8f;
typedef __attribute__((ext_vector_type(4)))  uint32_t v4u;
typedef __attribute__((ext_vector_type(8)))  uint32_t v8u;

#define NTOK   (64 * 2048)      // 131072 tokens
#define KPAD   512              // 510 feature dims padded to 512
#define ODIM   256
#define FEATD  510
#define KT_N   16               // 512 / 32 k-steps
#define BSLAB  8192             // halves per kt slab: 16 ntiles * 32 lanes * 16

// ---------------------------------------------------------------------------
// Kernel 1: pack W_out [256,510] f32 -> WMMA B fragments f16, kt-major so each
// kt slab (16 KB) is contiguous for the TDM:
//   Bpack[(((kt*16 + nt)*32 + lane)*16 + e]
//   lane<16 : N = nt*16+lane,    K = kt*32 + (e<8 ? e : 16+e-8)
//   lane>=16: N = nt*16+lane-16, K = kt*32 + 8 + (e<8 ? e : 16+e-8)
// ---------------------------------------------------------------------------
__global__ __launch_bounds__(256) void packB_kernel(
    const float* __restrict__ Wout, _Float16* __restrict__ Bpack)
{
    int id   = blockIdx.x * 256 + threadIdx.x;   // 0 .. 512*256-1
    int e    = id & 15;
    int lane = (id >> 4) & 31;
    int nt   = (id >> 9) & 15;
    int kt   = id >> 13;
    int n    = nt * 16 + (lane & 15);
    int k    = kt * 32 + ((e & 8) ? 16 : 0) + ((lane >> 4) * 8) + (e & 7);
    float v  = (k < FEATD) ? Wout[(size_t)n * FEATD + k] : 0.0f;
    Bpack[id] = (_Float16)v;
}

// ---------------------------------------------------------------------------
// Kernel 2: fused part-MLPs + graph + keys + LayerNorm -> f16 features.
// One thread per token; 32-thread (1 wave) blocks; per-thread s[255] in LDS.
// ---------------------------------------------------------------------------
__device__ __forceinline__ float keyval(const float* xs, const float* WK, int f)
{
    int j = f / 30, m = f - j * 30;                       // keys[j][m]
    return xs[2 * j] * WK[2 * m] + xs[2 * j + 1] * WK[2 * m + 1];
}

#define W_OFF_WL   0
#define W_OFF_WL1  96
#define W_OFF_BL1  912
#define W_OFF_WT   963
#define W_OFF_WT1  1123
#define W_OFF_BT1  1939
#define W_OFF_WK   1990
#define W_OFF_EMB  2050
#define W_OFF_WC   2305
#define W_OFF_BC   2356
#define W_OFF_LNA  2357
#define W_OFF_LNB  2867
#define W_TOTAL    3377

__global__ __launch_bounds__(32) void feat_kernel(
    const float* __restrict__ x,
    const float* __restrict__ Wl,  const float* __restrict__ Wl1,
    const float* __restrict__ bl1, const float* __restrict__ Wt,
    const float* __restrict__ Wt1, const float* __restrict__ bt1,
    const float* __restrict__ Wk,  const float* __restrict__ emb,
    const float* __restrict__ Wc,  const float* __restrict__ bc,
    const float* __restrict__ lna, const float* __restrict__ lnb,
    _Float16* __restrict__ featOut)
{
    __shared__ float wlds[W_TOTAL];
    __shared__ float slds[32 * 255];   // per-thread s[5][51]
    __shared__ float xlds[32 * 35];    // per-thread x (stride 35: conflict-free)

    const int tid = threadIdx.x;

    // cooperative weight staging
    for (int i = tid; i < 96;  i += 32) wlds[W_OFF_WL  + i] = Wl[i];
    for (int i = tid; i < 816; i += 32) wlds[W_OFF_WL1 + i] = Wl1[i];
    for (int i = tid; i < 51;  i += 32) wlds[W_OFF_BL1 + i] = bl1[i];
    for (int i = tid; i < 160; i += 32) wlds[W_OFF_WT  + i] = Wt[i];
    for (int i = tid; i < 816; i += 32) wlds[W_OFF_WT1 + i] = Wt1[i];
    for (int i = tid; i < 51;  i += 32) wlds[W_OFF_BT1 + i] = bt1[i];
    for (int i = tid; i < 60;  i += 32) wlds[W_OFF_WK  + i] = Wk[i];
    for (int i = tid; i < 255; i += 32) wlds[W_OFF_EMB + i] = emb[i];
    for (int i = tid; i < 51;  i += 32) wlds[W_OFF_WC  + i] = Wc[i];
    if (tid == 0) wlds[W_OFF_BC] = bc[0];
    for (int i = tid; i < 510; i += 32) wlds[W_OFF_LNA + i] = lna[i];
    for (int i = tid; i < 510; i += 32) wlds[W_OFF_LNB + i] = lnb[i];
    __syncthreads();

    const float* WL  = wlds + W_OFF_WL;
    const float* WL1 = wlds + W_OFF_WL1;
    const float* BL1 = wlds + W_OFF_BL1;
    const float* WT  = wlds + W_OFF_WT;
    const float* WT1 = wlds + W_OFF_WT1;
    const float* BT1 = wlds + W_OFF_BT1;
    const float* WK  = wlds + W_OFF_WK;
    const float* EMB = wlds + W_OFF_EMB;
    const float* WC  = wlds + W_OFF_WC;
    const float  BCV = wlds[W_OFF_BC];
    const float* LNA = wlds + W_OFF_LNA;
    const float* LNB = wlds + W_OFF_LNB;

    const int token = blockIdx.x * 32 + tid;
    float xr[34];
    #pragma unroll
    for (int i = 0; i < 34; ++i) xr[i] = x[(size_t)token * 34 + i];
    float* xs = xlds + tid * 35;
    #pragma unroll
    for (int i = 0; i < 34; ++i) xs[i] = xr[i];

    float* S = slds + tid * 255;       // s[part*51 + c]

    // ----- 4 limb encoders: 6 -> relu(16) -> tanh(51) + embed -----
    const int LIMB0[4] = {11, 4, 14, 1};
    const int LIMB1[4] = {12, 5, 15, 2};
    const int LIMB2[4] = {13, 6, 16, 3};
    #pragma unroll
    for (int p = 0; p < 4; ++p) {
        float in6[6];
        in6[0] = xr[2 * LIMB0[p]]; in6[1] = xr[2 * LIMB0[p] + 1];
        in6[2] = xr[2 * LIMB1[p]]; in6[3] = xr[2 * LIMB1[p] + 1];
        in6[4] = xr[2 * LIMB2[p]]; in6[5] = xr[2 * LIMB2[p] + 1];
        float h[16];
        #pragma unroll
        for (int o = 0; o < 16; ++o) {
            float acc = 0.0f;
            #pragma unroll
            for (int i = 0; i < 6; ++i) acc += WL[o * 6 + i] * in6[i];
            h[o] = fmaxf(acc, 0.0f);
        }
        for (int c = 0; c < 51; ++c) {
            float acc = BL1[c];
            #pragma unroll
            for (int i = 0; i < 16; ++i) acc += WL1[c * 16 + i] * h[i];
            S[p * 51 + c] = tanhf(acc) + EMB[p * 51 + c];
        }
    }
    // ----- trunk encoder: 10 -> relu(16) -> tanh(51) + embed -----
    {
        const int TR[5] = {0, 7, 8, 9, 10};
        float in10[10];
        #pragma unroll
        for (int q = 0; q < 5; ++q) {
            in10[2 * q]     = xr[2 * TR[q]];
            in10[2 * q + 1] = xr[2 * TR[q] + 1];
        }
        float h[16];
        #pragma unroll
        for (int o = 0; o < 16; ++o) {
            float acc = 0.0f;
            #pragma unroll
            for (int i = 0; i < 10; ++i) acc += WT[o * 10 + i] * in10[i];
            h[o] = fmaxf(acc, 0.0f);
        }
        for (int c = 0; c < 51; ++c) {
            float acc = BT1[c];
            #pragma unroll
            for (int i = 0; i < 16; ++i) acc += WT1[c * 16 + i] * h[i];
            S[4 * 51 + c] = tanhf(acc) + EMB[4 * 51 + c];
        }
    }

    // ----- adjacency: A[i][j] = sigmoid(d_i + d_j + b), diag forced to 1 -----
    float d[5];
    #pragma unroll
    for (int i = 0; i < 5; ++i) {
        float acc = 0.0f;
        for (int c = 0; c < 51; ++c) acc += S[i * 51 + c] * WC[c];
        d[i] = acc;
    }
    float A[5][5];
    #pragma unroll
    for (int i = 0; i < 5; ++i)
        #pragma unroll
        for (int j = 0; j < 5; ++j)
            A[i][j] = (i == j) ? 1.0f
                               : 1.0f / (1.0f + expf(-(d[i] + d[j] + BCV)));

    // ----- pass 1: statistics over feat[510] (layout: per part [g(51),s(51)]) -----
    float sum = 0.0f, sumsq = 0.0f;
    #pragma unroll
    for (int i = 0; i < 5; ++i) {
        for (int c = 0; c < 51; ++c) {
            float sv = S[i * 51 + c];
            float g = -sv;
            #pragma unroll
            for (int j = 0; j < 5; ++j) g += A[i][j] * S[j * 51 + c];
            g = fmaxf(g, 0.0f);
            int fa_i = i * 102 + c;
            int fb_i = fa_i + 51;
            float fa = g  + keyval(xs, WK, fa_i);
            float fb = sv + keyval(xs, WK, fb_i);
            sum   += fa + fb;
            sumsq += fa * fa + fb * fb;
        }
    }
    float mean = sum * (1.0f / 510.0f);
    float var  = fmaxf((sumsq - sum * sum * (1.0f / 510.0f)) * (1.0f / 509.0f), 0.0f);
    float inv  = 1.0f / (sqrtf(var) + 1e-6f);   // torch: eps added to unbiased std

    // ----- pass 2: recompute, normalize, emit f16 row (padded to 512) -----
    _Float16* orow = featOut + (size_t)token * KPAD;
    #pragma unroll
    for (int i = 0; i < 5; ++i) {
        for (int c = 0; c < 51; ++c) {
            float sv = S[i * 51 + c];
            float g = -sv;
            #pragma unroll
            for (int j = 0; j < 5; ++j) g += A[i][j] * S[j * 51 + c];
            g = fmaxf(g, 0.0f);
            int fa_i = i * 102 + c;
            int fb_i = fa_i + 51;
            float fa = g  + keyval(xs, WK, fa_i);
            float fb = sv + keyval(xs, WK, fb_i);
            float na = LNA[fa_i] * (fa - mean) * inv + LNB[fa_i];
            float nb = LNA[fb_i] * (fb - mean) * inv + LNB[fb_i];
            orow[fa_i] = (_Float16)na;
            orow[fb_i] = (_Float16)nb;
        }
    }
    orow[510] = (_Float16)0.0f;
    orow[511] = (_Float16)0.0f;
}

// ---------------------------------------------------------------------------
// TDM helper: 1-D DMA of one 16 KB B slab (4096 x 4B elements) into LDS.
// D# group0: count=1, lds_addr, 57-bit global addr, type=2 ("image").
// D# group1: data_size=2 (4B), tensor_dim0 = tile_dim0 = 4096, 1-D tile.
// ---------------------------------------------------------------------------
__device__ __forceinline__ void tdm_load_slab(uint32_t lds_off, const _Float16* gsrc)
{
    uint64_t ga = (uint64_t)(uintptr_t)gsrc;
    v4u g0;
    g0[0] = 1u;                                            // count=1, user mode
    g0[1] = lds_off;                                       // LDS byte address
    g0[2] = (uint32_t)ga;                                  // global_addr[31:0]
    g0[3] = (uint32_t)((ga >> 32) & 0x1FFFFFFu) | (2u << 30); // addr[56:32] | type=2
    v8u g1;
    g1[0] = 2u << 16;                                      // data_size = 4B
    g1[1] = 4096u << 16;                                   // tensor_dim0[15:0]
    g1[2] = (4096u >> 16) | (1u << 16);                    // dim0 hi | tensor_dim1=1
    g1[3] = 4096u << 16;                                   // tile_dim0 = 4096
    g1[4] = 0u;                                            // tile_dim1/2 unused
    g1[5] = 4096u;                                         // tensor_dim0_stride
    g1[6] = 0u;
    g1[7] = 0u;
    asm volatile("tensor_load_to_lds %0, %1" :: "s"(g0), "s"(g1) : "memory");
}

// ---------------------------------------------------------------------------
// Kernel 3: out[N,256] = feat_f16[N,512] @ Wout_f16^T via v_wmma_f32_16x16x32_f16.
// Block tile 128(M) x 256(N): 8 waves, each one 16x256 strip (16 accumulators).
// B slabs (16 KB per kt) staged LDS-resident via TDM, double-buffered; wave 0
// drives the DMA (TENSORcnt) and workgroup barriers publish the buffer.
// A is loaded straight from the row-major f16 feature buffer (the documented
// 16-bit A 16x32 layout = two contiguous 8-half chunks per lane).
// ---------------------------------------------------------------------------
__global__ __launch_bounds__(256) void gemm_kernel(
    const _Float16* __restrict__ Afeat,
    const _Float16* __restrict__ Bpack,
    float* __restrict__ out)
{
    __shared__ __align__(1024) _Float16 bbuf[2][BSLAB];    // 2 x 16 KB

    const int lane = threadIdx.x & 31;
    const int wid  = threadIdx.x >> 5;                     // 0..7 -> M strip
    const int half = lane >> 4;
    const int m_base = blockIdx.x * 128 + wid * 16;
    const int row    = m_base + (lane & 15);

    v8f acc[16];
    #pragma unroll
    for (int t = 0; t < 16; ++t) acc[t] = (v8f){};

    const _Float16* arow = Afeat + (size_t)row * KPAD + half * 8;

    // prologue: DMA slab for kt=0 into buffer 0
    if (wid == 0)
        tdm_load_slab((uint32_t)(uintptr_t)&bbuf[0][0], Bpack);

    for (int kt = 0; kt < KT_N; ++kt) {
        if (wid == 0)
            __builtin_amdgcn_s_wait_tensorcnt(0);          // slab kt arrived
        __syncthreads();                                   // publish LDS to all waves
        if (wid == 0 && kt + 1 < KT_N)                     // overlap next DMA
            tdm_load_slab((uint32_t)(uintptr_t)&bbuf[(kt + 1) & 1][0],
                          Bpack + (size_t)(kt + 1) * BSLAB);

        // A fragment: K = kt*32 + {half*8 .. +7} and {half*8+16 .. +23}
        const v8h a0 = *(const v8h*)(arow + kt * 32);
        const v8h a1 = *(const v8h*)(arow + kt * 32 + 16);
        v16h a;
        #pragma unroll
        for (int e = 0; e < 8; ++e) { a[e] = a0[e]; a[8 + e] = a1[e]; }

        const _Float16* bls = &bbuf[kt & 1][0];
        #pragma unroll
        for (int t = 0; t < 16; ++t) {
            const _Float16* bp = bls + t * 512 + lane * 16;
            const v8h b0 = *(const v8h*)bp;
            const v8h b1 = *(const v8h*)(bp + 8);
            v16h b;
            #pragma unroll
            for (int e = 0; e < 8; ++e) { b[e] = b0[e]; b[8 + e] = b1[e]; }

            acc[t] = __builtin_amdgcn_wmma_f32_16x16x32_f16(
                false, a, false, b, (short)0, acc[t], false, false);
        }
        __syncthreads();                                   // all reads done before overwrite
    }

    // C/D layout: VGPR r -> M = r + (lane<16 ? 0 : 8), N = lane & 15
    const int colbase = lane & 15;
    const int rbase   = m_base + half * 8;
    #pragma unroll
    for (int r = 0; r < 8; ++r) {
        float* op = out + (size_t)(rbase + r) * ODIM + colbase;
        #pragma unroll
        for (int t = 0; t < 16; ++t)
            op[t * 16] = acc[t][r];
    }
}

// ---------------------------------------------------------------------------
extern "C" void kernel_launch(void* const* d_in, const int* in_sizes, int n_in,
                              void* d_out, int out_size, void* d_ws, size_t ws_size,
                              hipStream_t stream)
{
    const float* x    = (const float*)d_in[0];
    const float* Wl   = (const float*)d_in[1];
    const float* Wl1  = (const float*)d_in[2];
    const float* bl1  = (const float*)d_in[3];
    const float* Wt   = (const float*)d_in[4];
    const float* Wt1  = (const float*)d_in[5];
    const float* bt1  = (const float*)d_in[6];
    const float* Wk   = (const float*)d_in[7];
    const float* emb  = (const float*)d_in[8];
    const float* Wc   = (const float*)d_in[9];
    const float* bc   = (const float*)d_in[10];
    const float* lna  = (const float*)d_in[11];
    const float* lnb  = (const float*)d_in[12];
    const float* Wout = (const float*)d_in[13];
    float* out = (float*)d_out;

    _Float16* Bpack = (_Float16*)d_ws;                                    // 256 KB
    _Float16* feat  = (_Float16*)((char*)d_ws + (size_t)KPAD * ODIM * 2); // 134 MB

    packB_kernel<<<(KPAD * ODIM) / 256, 256, 0, stream>>>(Wout, Bpack);
    feat_kernel<<<NTOK / 32, 32, 0, stream>>>(x, Wl, Wl1, bl1, Wt, Wt1, bt1,
                                              Wk, emb, Wc, bc, lna, lnb, feat);
    gemm_kernel<<<NTOK / 128, 256, 0, stream>>>(feat, Bpack, out);
}
